// rate_RNN_mnist_19980187861907
// MI455X (gfx1250) — matrix operand order repaired and checked by reading, not compiled, not run on hardware
//
#include <hip/hip_runtime.h>
#include <hip/hip_bf16.h>
#include <math.h>

// ---- problem constants (match reference) ----
#define TSTEPS 100
#define BATCH  256
#define INDIM  784
#define INPAD  800      // Win padded K stride (multiple of 32, rows 32B-aligned)
#define KFULL  768      // 24 full 32-wide chunks; tail chunk covers k 768..783
#define HID    1024
#define OUTD   10
#define PRANK  64
#define ALPHA  0.05f    // DT / TAUM

// LDS staging of Wr: 64 rows x 128 halves per slab, padded row stride 136
// (TDM pad: +4 DWORDs per 64-DWORD row -> 4-bank rotation, no conflicts)
#define SLAB_K   128
#define LDS_LDB  136

typedef __attribute__((ext_vector_type(16))) _Float16 v16h;
typedef __attribute__((ext_vector_type(8)))  _Float16 v8h;
typedef __attribute__((ext_vector_type(8)))  float    v8f;
typedef __attribute__((ext_vector_type(4)))  float    v4f;
typedef __attribute__((ext_vector_type(4)))  unsigned int v4u;
typedef __attribute__((ext_vector_type(8)))  int      v8i;
typedef __attribute__((ext_vector_type(4)))  int      v4i;

// ------------------------------------------------------------------
// WMMA fragment loaders (wave32, v_wmma_f32_16x16x32_f16).
// A (16x32 f16): lane m holds K = {kb..kb+7, 16+kb..16+kb+7}, kb=(lane&16)?8:0.
// B (32x16 f16): lane n = lane&15; same K split as A: {kb..kb+7, 16+kb..16+kb+7}.
// ------------------------------------------------------------------

__device__ __forceinline__ v16h ldA_h16(const _Float16* __restrict__ base, int ld) {
  const int lane = threadIdx.x & 31;
  const _Float16* p = base + (size_t)(lane & 15) * ld + ((lane & 16) ? 8 : 0);
  v8h lo = *(const v8h*)(p);
  v8h hi = *(const v8h*)(p + 16);
  return __builtin_shufflevector(lo, hi, 0, 1, 2, 3, 4, 5, 6, 7,
                                 8, 9, 10, 11, 12, 13, 14, 15);
}

__device__ __forceinline__ v16h ldA_f32_fast(const float* __restrict__ base, int ld) {
  const int lane = threadIdx.x & 31;
  const float* p = base + (size_t)(lane & 15) * ld + ((lane & 16) ? 8 : 0);
  v4f a0 = *(const v4f*)(p);
  v4f a1 = *(const v4f*)(p + 4);
  v4f a2 = *(const v4f*)(p + 16);
  v4f a3 = *(const v4f*)(p + 20);
  v16h f;
#pragma unroll
  for (int i = 0; i < 4; ++i) {
    f[i]      = (_Float16)a0[i];
    f[4 + i]  = (_Float16)a1[i];
    f[8 + i]  = (_Float16)a2[i];
    f[12 + i] = (_Float16)a3[i];
  }
  return f;
}

// tail chunk at k0=768: only k<16 valid (half-index i<8) -- compile-time guard
__device__ __forceinline__ v16h ldA_f32_tail(const float* __restrict__ base, int ld) {
  const int lane = threadIdx.x & 31;
  const float* p = base + (size_t)(lane & 15) * ld + ((lane & 16) ? 8 : 0);
  v4f a0 = *(const v4f*)(p);
  v4f a1 = *(const v4f*)(p + 4);
  v16h f;
#pragma unroll
  for (int i = 0; i < 4; ++i) {
    f[i]      = (_Float16)a0[i];
    f[4 + i]  = (_Float16)a1[i];
    f[8 + i]  = (_Float16)0.0f;
    f[12 + i] = (_Float16)0.0f;
  }
  return f;
}

__device__ __forceinline__ v16h ldB_h16(const _Float16* __restrict__ base, int ld) {
  const int lane = threadIdx.x & 31;
  const _Float16* p = base + (size_t)(lane & 15) * ld + ((lane & 16) ? 8 : 0);
  v8h lo = *(const v8h*)(p);
  v8h hi = *(const v8h*)(p + 16);
  return __builtin_shufflevector(lo, hi, 0, 1, 2, 3, 4, 5, 6, 7,
                                 8, 9, 10, 11, 12, 13, 14, 15);
}

// B fragment from the padded LDS slab (two ds_load_b128 per lane)
__device__ __forceinline__ v16h ldB_lds(const _Float16* base, int ld) {
  const int lane = threadIdx.x & 31;
  const _Float16* p = base + (lane & 15) * ld + ((lane & 16) ? 8 : 0);
  v8h lo = *(const v8h*)(p);
  v8h hi = *(const v8h*)(p + 16);
  return __builtin_shufflevector(lo, hi, 0, 1, 2, 3, 4, 5, 6, 7,
                                 8, 9, 10, 11, 12, 13, 14, 15);
}

#define WMMA_F16(A, B, C) __builtin_amdgcn_wmma_f32_16x16x32_f16( \
    false, (A), false, (B), (short)0, (C), false, false)

// ------------------------------------------------------------------
// Tensor Data Mover: DMA a 64x128 f16 tile (row pitch HID) from global
// into LDS with a 16-byte pad after each 256-byte row (bank-conflict free).
// D# packing per CDNA5 ISA.
// ------------------------------------------------------------------
#if __has_builtin(__builtin_amdgcn_tensor_load_to_lds)
#define USE_TDM 1
__device__ __forceinline__ void tdm_load_wr_slab(unsigned lds_off,
                                                 const _Float16* gsrc) {
  const unsigned long long ga = (unsigned long long)(size_t)gsrc;
  v4u g0;
  g0[0] = 1u;                                   // count=1, user descriptor
  g0[1] = lds_off;                              // lds_addr (bytes)
  g0[2] = (unsigned)ga;                         // global_addr[31:0]
  g0[3] = (unsigned)((ga >> 32) & 0x01FFFFFFu)  // global_addr[56:32]
          | (2u << 30);                         // type = 2 ("image")
  v8i g1;
  g1[0] = (1 << 16)                             // data_size = 2 bytes
        | (1 << 20)                             // pad_enable
        | (5 << 22)                             // pad_interval: 64 DWORDs (one row)
        | (3 << 25);                            // pad_amount: 4 DWORDs (16B)
  g1[1] = (SLAB_K << 16);                       // tensor_dim0[15:0] = 128
  g1[2] = (64 << 16);                           // dim0 hi=0 | tensor_dim1 lo = 64
  g1[3] = (SLAB_K << 16);                       // dim1 hi=0 | tile_dim0 = 128
  g1[4] = 64;                                   // tile_dim1 = 64 | tile_dim2 = 0
  g1[5] = HID;                                  // tensor_dim0_stride lo = 1024
  g1[6] = 0;
  g1[7] = 0;
  v4i z4 = {0, 0, 0, 0};
#if defined(__clang_major__) && (__clang_major__ >= 23)
  v8i z8 = {0, 0, 0, 0, 0, 0, 0, 0};
  __builtin_amdgcn_tensor_load_to_lds(g0, g1, z4, z4, z8, 0);
#else
  __builtin_amdgcn_tensor_load_to_lds(g0, g1, z4, z4, 0);
#endif
  __builtin_amdgcn_s_wait_tensorcnt(0);
}
#else
#define USE_TDM 0
#endif

// ------------------------------------------------------------------
// 0) init state
// ------------------------------------------------------------------
__global__ void init_kernel(float* __restrict__ mem, float* __restrict__ rm,
                            _Float16* __restrict__ tA, _Float16* __restrict__ tB) {
  const int idx = blockIdx.x * blockDim.x + threadIdx.x;   // < BATCH*HID
  mem[idx] = 0.0f;
  rm[idx]  = -3.4028235e38f;
  tA[idx]  = (_Float16)0.0f;
  tB[idx]  = (_Float16)0.0f;
}

// ------------------------------------------------------------------
// 1) Win f32[H][784] -> f16[H][800] zero-padded
// ------------------------------------------------------------------
__global__ void cvt_win_kernel(const float* __restrict__ Win, _Float16* __restrict__ Win16) {
  const int idx = blockIdx.x * blockDim.x + threadIdx.x;   // < HID*INPAD
  const int h = idx / INPAD, kp = idx - h * INPAD;
  Win16[idx] = (_Float16)((kp < INDIM) ? Win[(size_t)h * INDIM + kp] : 0.0f);
}

// ------------------------------------------------------------------
// 2) Wr[h][g] = sum_p pin[h][p] * l[p] * pout[g][p]
// ------------------------------------------------------------------
__global__ void wr_kernel(const float* __restrict__ pin, const float* __restrict__ pout,
                          const float* __restrict__ l, _Float16* __restrict__ Wr16) {
  const int idx = blockIdx.x * blockDim.x + threadIdx.x;   // < HID*HID
  const int h = idx >> 10, g = idx & (HID - 1);
  float acc = 0.0f;
#pragma unroll 16
  for (int p = 0; p < PRANK; ++p)
    acc += pin[(size_t)h * PRANK + p] * l[p] * pout[(size_t)g * PRANK + p];
  Wr16[idx] = (_Float16)acc;
}

// ------------------------------------------------------------------
// 3) hoisted input GEMM over ALL timesteps (M = 25600 rows)
// ------------------------------------------------------------------
__global__ __launch_bounds__(128) void xin_gemm_kernel(
    const float* __restrict__ spk,          // [M][784]
    const _Float16* __restrict__ Win16,     // [HID][INPAD]
    _Float16* __restrict__ Xin16)           // [M][HID]
{
  const int wave = threadIdx.x >> 5;
  const int lane = threadIdx.x & 31;
  const int m0 = blockIdx.y * 64 + wave * 16;
  const int n0 = blockIdx.x * 64;

  v8f acc[4] = {};
  for (int k0 = 0; k0 < KFULL; k0 += 32) {
    const v16h a = ldA_f32_fast(spk + (size_t)m0 * INDIM + k0, INDIM);
#pragma unroll
    for (int j = 0; j < 4; ++j) {
      const v16h b = ldB_h16(Win16 + (size_t)(n0 + 16 * j) * INPAD + k0, INPAD);
      acc[j] = WMMA_F16(a, b, acc[j]);
    }
  }
  {  // tail chunk, compile-time masked
    const v16h a = ldA_f32_tail(spk + (size_t)m0 * INDIM + KFULL, INDIM);
#pragma unroll
    for (int j = 0; j < 4; ++j) {
      const v16h b = ldB_h16(Win16 + (size_t)(n0 + 16 * j) * INPAD + KFULL, INPAD);
      acc[j] = WMMA_F16(a, b, acc[j]);
    }
  }

  const int ml = (lane & 16) ? 8 : 0;
  const int nl = lane & 15;
#pragma unroll
  for (int j = 0; j < 4; ++j)
#pragma unroll
    for (int r = 0; r < 8; ++r)
      Xin16[(size_t)(m0 + ml + r) * HID + n0 + 16 * j + nl] = (_Float16)acc[j][r];
}

// ------------------------------------------------------------------
// 4a) per-step kernel, hoisted-Xin path: recurrent GEMM with the Wr
//     strip staged in LDS via TDM (or cooperative fill fallback).
// ------------------------------------------------------------------
__global__ __launch_bounds__(128) void rnn_step_pre_kernel(
    const _Float16* __restrict__ Xin_t,     // [BATCH][HID]
    const _Float16* __restrict__ Wr16,      // [HID][HID]
    float* __restrict__ mem,
    const _Float16* __restrict__ tin,
    _Float16* __restrict__ tout,
    float* __restrict__ rm)
{
  __shared__ _Float16 bsh[64 * LDS_LDB];    // 17 KB padded Wr slab

  const int wave = threadIdx.x >> 5;
  const int lane = threadIdx.x & 31;
  const int m0 = blockIdx.y * 64 + wave * 16;
  const int n0 = blockIdx.x * 64;

  v8f acc[4] = {};
  for (int ks = 0; ks < HID; ks += SLAB_K) {
    __syncthreads();                        // previous slab fully consumed
    const _Float16* gsrc = Wr16 + (size_t)n0 * HID + ks;
#if USE_TDM
    if (threadIdx.x < 32) {                 // one DMA per block, wave 0 issues
      tdm_load_wr_slab((unsigned)(size_t)&bsh[0], gsrc);
    }
#else
    for (int i = threadIdx.x; i < 64 * (SLAB_K / 8); i += 128) {
      const int row = i >> 4, c8 = (i & 15) * 8;
      *(v8h*)(bsh + row * LDS_LDB + c8) = *(const v8h*)(gsrc + (size_t)row * HID + c8);
    }
#endif
    __syncthreads();                        // slab visible to all waves

#pragma unroll
    for (int kk = 0; kk < SLAB_K; kk += 32) {
      const v16h a = ldA_h16(tin + (size_t)m0 * HID + ks + kk, HID);
#pragma unroll
      for (int j = 0; j < 4; ++j) {
        const v16h b = ldB_lds(bsh + (16 * j) * LDS_LDB + kk, LDS_LDB);
        acc[j] = WMMA_F16(a, b, acc[j]);
      }
    }
  }

  const int ml = (lane & 16) ? 8 : 0;
  const int nl = lane & 15;
#pragma unroll
  for (int j = 0; j < 4; ++j) {
#pragma unroll
    for (int r = 0; r < 8; ++r) {
      const size_t idx = (size_t)(m0 + ml + r) * HID + n0 + 16 * j + nl;
      const float I    = acc[j][r] + (float)Xin_t[idx];
      const float newm = (I - mem[idx]) * ALPHA;
      const float th   = tanhf(newm);
      mem[idx]  = newm;
      tout[idx] = (_Float16)th;
      rm[idx]   = fmaxf(rm[idx], th);
    }
  }
}

// ------------------------------------------------------------------
// 4b) per-step kernel, fused fallback (small ws): both GEMMs in-step
// ------------------------------------------------------------------
__global__ __launch_bounds__(128) void rnn_step_fused_kernel(
    const float* __restrict__ spk_t,
    const _Float16* __restrict__ Win16,
    const _Float16* __restrict__ Wr16,
    float* __restrict__ mem,
    const _Float16* __restrict__ tin,
    _Float16* __restrict__ tout,
    float* __restrict__ rm)
{
  const int wave = threadIdx.x >> 5;
  const int lane = threadIdx.x & 31;
  const int m0 = blockIdx.y * 64 + wave * 16;
  const int n0 = blockIdx.x * 64;

  v8f acc[4] = {};
  for (int k0 = 0; k0 < KFULL; k0 += 32) {
    const v16h a = ldA_f32_fast(spk_t + (size_t)m0 * INDIM + k0, INDIM);
#pragma unroll
    for (int j = 0; j < 4; ++j) {
      const v16h b = ldB_h16(Win16 + (size_t)(n0 + 16 * j) * INPAD + k0, INPAD);
      acc[j] = WMMA_F16(a, b, acc[j]);
    }
  }
  {
    const v16h a = ldA_f32_tail(spk_t + (size_t)m0 * INDIM + KFULL, INDIM);
#pragma unroll
    for (int j = 0; j < 4; ++j) {
      const v16h b = ldB_h16(Win16 + (size_t)(n0 + 16 * j) * INPAD + KFULL, INPAD);
      acc[j] = WMMA_F16(a, b, acc[j]);
    }
  }
  for (int k0 = 0; k0 < HID; k0 += 32) {
    const v16h a = ldA_h16(tin + (size_t)m0 * HID + k0, HID);
#pragma unroll
    for (int j = 0; j < 4; ++j) {
      const v16h b = ldB_h16(Wr16 + (size_t)(n0 + 16 * j) * HID + k0, HID);
      acc[j] = WMMA_F16(a, b, acc[j]);
    }
  }

  const int ml = (lane & 16) ? 8 : 0;
  const int nl = lane & 15;
#pragma unroll
  for (int j = 0; j < 4; ++j) {
#pragma unroll
    for (int r = 0; r < 8; ++r) {
      const size_t idx = (size_t)(m0 + ml + r) * HID + n0 + 16 * j + nl;
      const float newm = (acc[j][r] - mem[idx]) * ALPHA;
      const float th   = tanhf(newm);
      mem[idx]  = newm;
      tout[idx] = (_Float16)th;
      rm[idx]   = fmaxf(rm[idx], th);
    }
  }
}

// ------------------------------------------------------------------
// 5) readout: Rout = rm @ Wout^T, softmax. One wave per batch row.
// ------------------------------------------------------------------
__global__ __launch_bounds__(32) void readout_kernel(
    const float* __restrict__ rm, const float* __restrict__ Wout,
    float* __restrict__ out)
{
  const int b = blockIdx.x;
  const int lane = threadIdx.x;
  float acc[OUTD];
#pragma unroll
  for (int o = 0; o < OUTD; ++o) acc[o] = 0.0f;
  for (int h = lane; h < HID; h += 32) {
    const float r = rm[(size_t)b * HID + h];
#pragma unroll
    for (int o = 0; o < OUTD; ++o) acc[o] += r * Wout[(size_t)o * HID + h];
  }
#pragma unroll
  for (int o = 0; o < OUTD; ++o)
#pragma unroll
    for (int off = 16; off > 0; off >>= 1)
      acc[o] += __shfl_xor(acc[o], off, 32);
  if (lane == 0) {
    float mx = acc[0];
#pragma unroll
    for (int o = 1; o < OUTD; ++o) mx = fmaxf(mx, acc[o]);
    float e[OUTD], s = 0.0f;
#pragma unroll
    for (int o = 0; o < OUTD; ++o) { e[o] = expf(acc[o] - mx); s += e[o]; }
    const float inv = 1.0f / s;
#pragma unroll
    for (int o = 0; o < OUTD; ++o) out[(size_t)b * OUTD + o] = e[o] * inv;
  }
}

// ------------------------------------------------------------------
// host-side launch
// ------------------------------------------------------------------
extern "C" void kernel_launch(void* const* d_in, const int* in_sizes, int n_in,
                              void* d_out, int out_size, void* d_ws, size_t ws_size,
                              hipStream_t stream) {
  (void)in_sizes; (void)n_in; (void)out_size;
  const float* spk  = (const float*)d_in[0];  // [T][B][784]
  const float* Win  = (const float*)d_in[1];  // [1024][784]
  const float* Wout = (const float*)d_in[2];  // [10][1024]
  const float* pin  = (const float*)d_in[3];  // [1024][64]
  const float* pout = (const float*)d_in[4];  // [1024][64]
  const float* l    = (const float*)d_in[5];  // [64]
  float* out = (float*)d_out;                 // [256][10]

  char* w = (char*)d_ws;
  auto carve = [&](size_t bytes) -> char* {
    char* p = w;
    w += (bytes + 255) & ~(size_t)255;
    return p;
  };
  _Float16* Wr16  = (_Float16*)carve((size_t)HID * HID * sizeof(_Float16));
  _Float16* Win16 = (_Float16*)carve((size_t)HID * INPAD * sizeof(_Float16));
  float*    mem   = (float*)carve((size_t)BATCH * HID * sizeof(float));
  float*    rm    = (float*)carve((size_t)BATCH * HID * sizeof(float));
  _Float16* tA    = (_Float16*)carve((size_t)BATCH * HID * sizeof(_Float16));
  _Float16* tB    = (_Float16*)carve((size_t)BATCH * HID * sizeof(_Float16));

  const size_t xin_bytes = (size_t)TSTEPS * BATCH * HID * sizeof(_Float16);
  const size_t used = (size_t)(w - (char*)d_ws);
  const bool hoist = (ws_size >= used + xin_bytes + 256);
  _Float16* Xin16 = hoist ? (_Float16*)carve(xin_bytes) : nullptr;

  init_kernel<<<(BATCH * HID) / 256, 256, 0, stream>>>(mem, rm, tA, tB);
  cvt_win_kernel<<<(HID * INPAD) / 256, 256, 0, stream>>>(Win, Win16);
  wr_kernel<<<(HID * HID) / 256, 256, 0, stream>>>(pin, pout, l, Wr16);

  _Float16* tin = tA;
  _Float16* tout_ = tB;

  if (hoist) {
    xin_gemm_kernel<<<dim3(HID / 64, (TSTEPS * BATCH) / 64), 128, 0, stream>>>(
        spk, Win16, Xin16);
    for (int t = 0; t < TSTEPS; ++t) {
      rnn_step_pre_kernel<<<dim3(HID / 64, BATCH / 64), 128, 0, stream>>>(
          Xin16 + (size_t)t * BATCH * HID, Wr16, mem, tin, tout_, rm);
      _Float16* tmp = tin; tin = tout_; tout_ = tmp;
    }
  } else {
    for (int t = 0; t < TSTEPS; ++t) {
      const float* spk_t = spk + (size_t)t * BATCH * INDIM;
      rnn_step_fused_kernel<<<dim3(HID / 64, BATCH / 64), 128, 0, stream>>>(
          spk_t, Win16, Wr16, mem, tin, tout_, rm);
      _Float16* tmp = tin; tin = tout_; tout_ = tmp;
    }
  }

  readout_kernel<<<BATCH, 32, 0, stream>>>(rm, Wout, out);
}
